// SplitTrans_58428735094816
// MI455X (gfx1250) — matrix operand order repaired and checked by reading, not compiled, not run on hardware
//
#include <hip/hip_runtime.h>

typedef unsigned short u16;
typedef unsigned int u32;
typedef __attribute__((ext_vector_type(16))) __bf16 v16bf;
typedef __attribute__((ext_vector_type(8)))  __bf16 v8bf;
typedef __attribute__((ext_vector_type(8)))  float  v8f;

constexpr int kB = 16, kN = 1024, kD = 512, kH = 8, kHD = 64, kDFF = 2048, kNG = 4;
constexpr float kNEG = -1e9f;

// ---------------- bf16 helpers ----------------
__device__ __forceinline__ u16 f2bf(float f) {
    union { float f; u32 u; } v; v.f = f;
    u32 u = v.u + 0x7FFFu + ((v.u >> 16) & 1u);
    return (u16)(u >> 16);
}

// ---------------- async global->LDS copy (CDNA5 ASYNCcnt path) ----------------
// Copies 64 contiguous bytes; INST_OFFSET is added to both the LDS and global
// addresses (ISA 08_async_tensor pseudo-code), so one base covers 4 x b128.
__device__ __forceinline__ void async_copy64B(u32 lds_byte, const u16* gptr) {
    unsigned long long ga = (unsigned long long)(uintptr_t)gptr;
    asm volatile(
        "global_load_async_to_lds_b128 %0, %1, off\n\t"
        "global_load_async_to_lds_b128 %0, %1, off offset:16\n\t"
        "global_load_async_to_lds_b128 %0, %1, off offset:32\n\t"
        "global_load_async_to_lds_b128 %0, %1, off offset:48"
        :: "v"(lds_byte), "v"(ga) : "memory");
}
__device__ __forceinline__ void async_copy32B(u32 lds_byte, const u16* gptr) {
    unsigned long long ga = (unsigned long long)(uintptr_t)gptr;
    asm volatile(
        "global_load_async_to_lds_b128 %0, %1, off\n\t"
        "global_load_async_to_lds_b128 %0, %1, off offset:16"
        :: "v"(lds_byte), "v"(ga) : "memory");
}
__device__ __forceinline__ void wait_async0() {
    asm volatile("s_wait_asynccnt 0" ::: "memory");
}

// ---------------- WMMA fragment loaders (16-bit A/B layouts, ISA 7.12.2) ----
// A (16x32): lane l -> row M=l&15, K chunks [half*8..half*8+7] and [16+half*8..]
__device__ __forceinline__ v16bf load_a_frag(const u16* base, int ld, int lane) {
    int half = lane >> 4, m = lane & 15;
    v8bf lo = *(const v8bf*)(base + m * ld + half * 8);
    v8bf hi = *(const v8bf*)(base + m * ld + 16 + half * 8);
    v16bf a;
#pragma unroll
    for (int i = 0; i < 8; ++i) { a[i] = lo[i]; a[i + 8] = hi[i]; }
    return a;
}
// B (32x16): lane l -> col N=l&15, K chunks [half*8..half*8+7] and [16+half*8..]
__device__ __forceinline__ v16bf load_b_frag(const u16* base, int ld, int lane) {
    int half = lane >> 4, n = lane & 15;
    v8bf lo = *(const v8bf*)(base + n * ld + half * 8);
    v8bf hi = *(const v8bf*)(base + n * ld + 16 + half * 8);
    v16bf b;
#pragma unroll
    for (int i = 0; i < 8; ++i) { b[i] = lo[i]; b[i + 8] = hi[i]; }
    return b;
}

// ---------------- small utility kernels ----------------
__global__ void copy_f4_kernel(const float4* __restrict__ src, float4* __restrict__ dst, int n4) {
    int i = blockIdx.x * blockDim.x + threadIdx.x, s = gridDim.x * blockDim.x;
    for (int j = i; j < n4; j += s) dst[j] = src[j];
}
__global__ void zero_kernel(float4* a, int n4, uint4* b, int n4b) {
    int i = blockIdx.x * blockDim.x + threadIdx.x, s = gridDim.x * blockDim.x;
    float4 zf = make_float4(0.f, 0.f, 0.f, 0.f);
    uint4  zu = make_uint4(0u, 0u, 0u, 0u);
    for (int j = i; j < n4;  j += s) a[j] = zf;
    for (int j = i; j < n4b; j += s) b[j] = zu;
}
__global__ void cvt_bf16_kernel(const float* __restrict__ src, u16* __restrict__ dst, int n) {
    int i = blockIdx.x * blockDim.x + threadIdx.x, s = gridDim.x * blockDim.x;
    for (int j = i; j < n; j += s) dst[j] = f2bf(src[j]);
}
__global__ void reset_kernel(int* p) { if (threadIdx.x == 0) p[0] = 0; }

// Per-batch prefix scan of the group mask -> packed rank (-1 if unselected),
// atomicMax of per-batch count into max_node.
__global__ __launch_bounds__(1024)
void scan_kernel(const int* __restrict__ cluster, const unsigned char* __restrict__ deg,
                 int g, int* __restrict__ rank, int* __restrict__ maxnode) {
    __shared__ int sm[1024];
    int b = blockIdx.x, tid = threadIdx.x;
    int idx = b * kN + tid;
    int mi = (deg[idx] != 0 && cluster[idx] == g) ? 1 : 0;
    sm[tid] = mi;
    __syncthreads();
    for (int off = 1; off < 1024; off <<= 1) {
        int t = (tid >= off) ? sm[tid - off] : 0;
        __syncthreads();
        sm[tid] += t;
        __syncthreads();
    }
    rank[idx] = mi ? (sm[tid] - 1) : -1;
    if (tid == 1023) atomicMax(maxnode, sm[1023]);
}

// Scatter selected rows into packed buffers (fp32 + bf16 shadow).
__global__ __launch_bounds__(128)
void pack_kernel(const float* __restrict__ x, const int* __restrict__ rank,
                 float* __restrict__ xp, u16* __restrict__ xpb) {
    int t = blockIdx.x;
    int r = rank[t];
    if (r < 0) return;
    int b = t >> 10;
    size_t src = (size_t)t * kD + threadIdx.x * 4;
    size_t dst = ((size_t)(b * kN + r)) * kD + threadIdx.x * 4;
    float4 v = *(const float4*)(x + src);
    *(float4*)(xp + dst) = v;
    u32 lo = (u32)f2bf(v.x) | ((u32)f2bf(v.y) << 16);
    u32 hi = (u32)f2bf(v.z) | ((u32)f2bf(v.w) << 16);
    *(uint2*)(xpb + dst) = make_uint2(lo, hi);
}

__global__ __launch_bounds__(128)
void scatter_kernel(const float* __restrict__ y, const int* __restrict__ rank,
                    float* __restrict__ out) {
    int t = blockIdx.x;
    int r = rank[t];
    if (r < 0) return;
    int b = t >> 10;
    size_t dst = (size_t)t * kD + threadIdx.x * 4;
    size_t src = ((size_t)(b * kN + r)) * kD + threadIdx.x * 4;
    *(float4*)(out + dst) = *(const float4*)(y + src);
}

// Residual add + LayerNorm; fp32 out (+ optional bf16 shadow).
__global__ __launch_bounds__(128)
void ln_kernel(const float* __restrict__ res, const float* __restrict__ add,
               const float* __restrict__ gamma, const float* __restrict__ beta,
               float* __restrict__ outF, u16* __restrict__ outB) {
    __shared__ float sbuf[8];
    int row = blockIdx.x, tid = threadIdx.x;
    size_t base = (size_t)row * kD + tid * 4;
    float4 a = *(const float4*)(res + base);
    float4 c = *(const float4*)(add + base);
    float v0 = a.x + c.x, v1 = a.y + c.y, v2 = a.z + c.z, v3 = a.w + c.w;
    float s1 = v0 + v1 + v2 + v3;
    float s2 = v0 * v0 + v1 * v1 + v2 * v2 + v3 * v3;
#pragma unroll
    for (int m = 1; m < 32; m <<= 1) { s1 += __shfl_xor(s1, m, 32); s2 += __shfl_xor(s2, m, 32); }
    int w = tid >> 5, lane = tid & 31;
    if (lane == 0) { sbuf[w] = s1; sbuf[4 + w] = s2; }
    __syncthreads();
    s1 = sbuf[0] + sbuf[1] + sbuf[2] + sbuf[3];
    s2 = sbuf[4] + sbuf[5] + sbuf[6] + sbuf[7];
    float mu = s1 * (1.f / kD);
    float var = s2 * (1.f / kD) - mu * mu;
    float rs = rsqrtf(var + 1e-5f);
    int c0 = tid * 4;
    float y0 = (v0 - mu) * rs * gamma[c0 + 0] + beta[c0 + 0];
    float y1 = (v1 - mu) * rs * gamma[c0 + 1] + beta[c0 + 1];
    float y2 = (v2 - mu) * rs * gamma[c0 + 2] + beta[c0 + 2];
    float y3 = (v3 - mu) * rs * gamma[c0 + 3] + beta[c0 + 3];
    *(float4*)(outF + base) = make_float4(y0, y1, y2, y3);
    if (outB) {
        u32 lo = (u32)f2bf(y0) | ((u32)f2bf(y1) << 16);
        u32 hi = (u32)f2bf(y2) | ((u32)f2bf(y3) << 16);
        *(uint2*)(outB + base) = make_uint2(lo, hi);
    }
}

// ---------------- WMMA GEMM: C[M,N] = A[M,K] * W[N,K]^T + bias ----------------
// 128x128x64 block tile, 8 waves (2x4), each wave 64x32 = 8 accumulators.
// Double-buffered LDS fed by async global->LDS copies (ASYNCcnt) so the next
// k-tile streams in while the current one runs through the WMMAs.
__global__ __launch_bounds__(256)
void gemm_bf16_kernel(const u16* __restrict__ A, const u16* __restrict__ W,
                      const float* __restrict__ bias, float* __restrict__ outF,
                      u16* __restrict__ outB, int M, int Nn, int K, int relu) {
    __shared__ __align__(32) u16 As[2][128 * 64];
    __shared__ __align__(32) u16 Bs[2][128 * 64];
    const int tid = threadIdx.x;
    const int lane = tid & 31, w = tid >> 5;
    const int wm = w >> 2, wn = w & 3;
    const int bm = blockIdx.y * 128, bn = blockIdx.x * 128;

    v8f acc[4][2];
#pragma unroll
    for (int i = 0; i < 4; ++i)
#pragma unroll
        for (int j = 0; j < 2; ++j)
#pragma unroll
            for (int e = 0; e < 8; ++e) acc[i][j][e] = 0.f;

    const int srow = tid >> 1;
    const int scol = (tid & 1) * 32;
    const u16* gA = A + (size_t)(bm + srow) * K + scol;
    const u16* gB = W + (size_t)(bn + srow) * K + scol;
    const u32 laddrA[2] = { (u32)(uintptr_t)&As[0][srow * 64 + scol],
                            (u32)(uintptr_t)&As[1][srow * 64 + scol] };
    const u32 laddrB[2] = { (u32)(uintptr_t)&Bs[0][srow * 64 + scol],
                            (u32)(uintptr_t)&Bs[1][srow * 64 + scol] };

    const int ktiles = K >> 6;
    // prologue: stage tile 0 into buffer 0
    async_copy64B(laddrA[0], gA);
    async_copy64B(laddrB[0], gB);

    for (int kt = 0; kt < ktiles; ++kt) {
        const int p = kt & 1;
        wait_async0();        // our buffer-p copies have landed
        __syncthreads();      // everyone's copies landed; prev reads of 1-p done
        if (kt + 1 < ktiles) {
            const int k1 = (kt + 1) << 6;
            async_copy64B(laddrA[1 - p], gA + k1);
            async_copy64B(laddrB[1 - p], gB + k1);
        }
        const u16* as = As[p];
        const u16* bs = Bs[p];
#pragma unroll
        for (int ks = 0; ks < 2; ++ks) {
            v16bf af[4], bfr[2];
#pragma unroll
            for (int mt = 0; mt < 4; ++mt)
                af[mt] = load_a_frag(as + (wm * 64 + mt * 16) * 64 + ks * 32, 64, lane);
#pragma unroll
            for (int nt = 0; nt < 2; ++nt)
                bfr[nt] = load_b_frag(bs + (wn * 32 + nt * 16) * 64 + ks * 32, 64, lane);
#pragma unroll
            for (int mt = 0; mt < 4; ++mt)
#pragma unroll
                for (int nt = 0; nt < 2; ++nt)
                    acc[mt][nt] = __builtin_amdgcn_wmma_f32_16x16x32_bf16(
                        false, af[mt], false, bfr[nt], (short)0, acc[mt][nt], false, false);
        }
    }
    const int half = lane >> 4, nl = lane & 15;
#pragma unroll
    for (int mt = 0; mt < 4; ++mt)
#pragma unroll
        for (int nt = 0; nt < 2; ++nt) {
            int gc = bn + wn * 32 + nt * 16 + nl;
            float bv = bias ? bias[gc] : 0.f;
#pragma unroll
            for (int r = 0; r < 8; ++r) {
                int gr = bm + wm * 64 + mt * 16 + r + 8 * half;
                float v = acc[mt][nt][r] + bv;
                if (relu) v = fmaxf(v, 0.f);
                size_t o = (size_t)gr * Nn + gc;
                if (outF) outF[o] = v;
                if (outB) outB[o] = f2bf(v);
            }
        }
}

// ---------------- Flash attention (per (b,h), 64 q-rows per block) ------------
__global__ __launch_bounds__(128)
void attn_kernel(const u16* __restrict__ qkv, u16* __restrict__ out,
                 const int* __restrict__ maxnode) {
    __shared__ __align__(32) u16 Ks[64 * 64];   // [key][hd]
    __shared__ __align__(32) u16 Vt[64 * 64];   // [hd][key]
    __shared__ __align__(32) u16 Ps[4 * 16 * 64];
    const int tid = threadIdx.x, lane = tid & 31, w = tid >> 5;
    const int bh = blockIdx.y;
    const int b = bh >> 3, h = bh & 7;
    const int qbase = blockIdx.x * 64 + w * 16;
    const int mn = maxnode[0];
    const int half = lane >> 4, nl = lane & 15;
    const int ld3 = 3 * kD;

    const u16* qrow = qkv + ((size_t)(b * kN + qbase)) * ld3 + h * kHD;
    v16bf qf[2];
    qf[0] = load_a_frag(qrow, ld3, lane);
    qf[1] = load_a_frag(qrow + 32, ld3, lane);

    const int srow = tid >> 1, shh = (tid & 1) * 32;
    const u32 lK = (u32)(uintptr_t)&Ks[srow * 64 + shh];

    float mrun[8], lrun[8];
    v8f oacc[4];
#pragma unroll
    for (int r = 0; r < 8; ++r) { mrun[r] = -1e30f; lrun[r] = 0.f; }
#pragma unroll
    for (int nt = 0; nt < 4; ++nt)
#pragma unroll
        for (int e = 0; e < 8; ++e) oacc[nt][e] = 0.f;

    for (int kb = 0; kb < kN / 64; ++kb) {
        __syncthreads();
        {   // K block: async direct-to-LDS; V block: transposed via scalar stores
            const u16* kg = qkv + ((size_t)(b * kN + kb * 64 + srow)) * ld3 + kD + h * kHD + shh;
            async_copy32B(lK, kg);
            const u16* vg = qkv + ((size_t)(b * kN + kb * 64 + srow)) * ld3 + 2 * kD + h * kHD + shh;
            union { uint4 q[4]; u16 s[32]; } tmp;
#pragma unroll
            for (int j = 0; j < 4; ++j) tmp.q[j] = ((const uint4*)vg)[j];
#pragma unroll
            for (int j = 0; j < 32; ++j) Vt[(shh + j) * 64 + srow] = tmp.s[j];
        }
        wait_async0();
        __syncthreads();

        // S = Q K^T
        v8f sacc[4];
#pragma unroll
        for (int nt = 0; nt < 4; ++nt) {
#pragma unroll
            for (int e = 0; e < 8; ++e) sacc[nt][e] = 0.f;
#pragma unroll
            for (int s = 0; s < 2; ++s) {
                v16bf bk = load_b_frag(Ks + (nt * 16) * 64 + s * 32, 64, lane);
                sacc[nt] = __builtin_amdgcn_wmma_f32_16x16x32_bf16(
                    false, qf[s], false, bk, (short)0, sacc[nt], false, false);
            }
        }
        // scale + key mask
#pragma unroll
        for (int nt = 0; nt < 4; ++nt) {
            int key = kb * 64 + nt * 16 + nl;
            float msk = (key < mn) ? 0.f : kNEG;
#pragma unroll
            for (int r = 0; r < 8; ++r) sacc[nt][r] = sacc[nt][r] * 0.125f + msk;
        }
        // online softmax (row reductions across the 16-lane half-groups)
#pragma unroll
        for (int r = 0; r < 8; ++r) {
            float mx = sacc[0][r];
#pragma unroll
            for (int nt = 1; nt < 4; ++nt) mx = fmaxf(mx, sacc[nt][r]);
#pragma unroll
            for (int m = 1; m < 16; m <<= 1) mx = fmaxf(mx, __shfl_xor(mx, m, 32));
            float mnew = fmaxf(mrun[r], mx);
            float alpha = __expf(mrun[r] - mnew);
            float rs = 0.f;
#pragma unroll
            for (int nt = 0; nt < 4; ++nt) {
                float p = __expf(sacc[nt][r] - mnew);
                rs += p;
                Ps[w * 1024 + (r + 8 * half) * 64 + nt * 16 + nl] = f2bf(p);
            }
#pragma unroll
            for (int m = 1; m < 16; m <<= 1) rs += __shfl_xor(rs, m, 32);
            lrun[r] = lrun[r] * alpha + rs;
            mrun[r] = mnew;
#pragma unroll
            for (int nt = 0; nt < 4; ++nt) oacc[nt][r] = oacc[nt][r] * alpha;
        }
        asm volatile("s_wait_dscnt 0" ::: "memory");
        // O += P V
        v16bf pf[2];
        pf[0] = load_a_frag(Ps + w * 1024, 64, lane);
        pf[1] = load_a_frag(Ps + w * 1024 + 32, 64, lane);
#pragma unroll
        for (int nt = 0; nt < 4; ++nt)
#pragma unroll
            for (int s = 0; s < 2; ++s) {
                v16bf bv = load_b_frag(Vt + (nt * 16) * 64 + s * 32, 64, lane);
                oacc[nt] = __builtin_amdgcn_wmma_f32_16x16x32_bf16(
                    false, pf[s], false, bv, (short)0, oacc[nt], false, false);
            }
    }
    // normalize + store bf16 attention output [b,n,h*64+hd]
#pragma unroll
    for (int nt = 0; nt < 4; ++nt)
#pragma unroll
        for (int r = 0; r < 8; ++r) {
            float li = lrun[r];
            float val = (li > 0.f) ? (oacc[nt][r] / li) : 0.f;
            int grow = b * kN + qbase + r + 8 * half;
            int gcol = h * kHD + nt * 16 + nl;
            out[(size_t)grow * kD + gcol] = f2bf(val);
        }
}

// ---------------- orchestration ----------------
extern "C" void kernel_launch(void* const* d_in, const int* in_sizes, int n_in,
                              void* d_out, int out_size, void* d_ws, size_t ws_size,
                              hipStream_t stream) {
    (void)in_sizes; (void)n_in; (void)out_size; (void)ws_size;
    const float* x          = (const float*)d_in[0];
    const int* cluster      = (const int*)d_in[1];
    const unsigned char* dm = (const unsigned char*)d_in[2];
    const float* Wqkv = (const float*)d_in[3];
    const float* bqkv = (const float*)d_in[4];
    const float* Wo   = (const float*)d_in[5];
    const float* bo   = (const float*)d_in[6];
    const float* W1   = (const float*)d_in[7];
    const float* b1   = (const float*)d_in[8];
    const float* W2   = (const float*)d_in[9];
    const float* b2   = (const float*)d_in[10];
    const float* g1   = (const float*)d_in[11];
    const float* be1  = (const float*)d_in[12];
    const float* g2   = (const float*)d_in[13];
    const float* be2  = (const float*)d_in[14];
    float* out = (float*)d_out;

    const size_t BND = (size_t)kB * kN * kD;           // 8,388,608
    char* ws = (char*)d_ws;
    size_t off = 0;
    auto alloc = [&](size_t bytes) {
        size_t cur = off;
        off = (off + bytes + 255) & ~(size_t)255;
        return cur;
    };
    int*   rank    = (int*)(ws + alloc((size_t)kB * kN * 4));
    int*   maxnode = (int*)(ws + alloc(256));
    float* bufA    = (float*)(ws + alloc(BND * 4));                 // xp fp32 / ffn out / final y
    u16*   bufB    = (u16*)(ws + alloc(BND * 2));                   // xp_bf / attn_bf / x1_bf
    u16*   bufC    = (u16*)(ws + alloc((size_t)kB * kN * kDFF * 2));// qkv_bf / h_bf
    float* bufD    = (float*)(ws + alloc(BND * 4));                 // o-proj out / x1 fp32
    u16*   wq_bf   = (u16*)(ws + alloc((size_t)3 * kD * kD * 2));
    u16*   wo_bf   = (u16*)(ws + alloc((size_t)kD * kD * 2));
    u16*   w1_bf   = (u16*)(ws + alloc((size_t)kDFF * kD * 2));
    u16*   w2_bf   = (u16*)(ws + alloc((size_t)kD * kDFF * 2));

    const int M = kB * kN;  // 16384

    // out = x
    copy_f4_kernel<<<4096, 256, 0, stream>>>((const float4*)x, (float4*)out, (int)(BND / 4));

    for (int g = 0; g < kNG; ++g) {
        // weights -> bf16
        cvt_bf16_kernel<<<2048, 256, 0, stream>>>(Wqkv + (size_t)g * 3 * kD * kD, wq_bf, 3 * kD * kD);
        cvt_bf16_kernel<<<1024, 256, 0, stream>>>(Wo + (size_t)g * kD * kD, wo_bf, kD * kD);
        cvt_bf16_kernel<<<2048, 256, 0, stream>>>(W1 + (size_t)g * kDFF * kD, w1_bf, kDFF * kD);
        cvt_bf16_kernel<<<2048, 256, 0, stream>>>(W2 + (size_t)g * kD * kDFF, w2_bf, kD * kDFF);

        reset_kernel<<<1, 32, 0, stream>>>(maxnode);
        scan_kernel<<<kB, 1024, 0, stream>>>(cluster, dm, g, rank, maxnode);
        zero_kernel<<<4096, 256, 0, stream>>>((float4*)bufA, (int)(BND / 4),
                                              (uint4*)bufB, (int)(BND * 2 / 16));
        pack_kernel<<<M, 128, 0, stream>>>(x, rank, bufA, bufB);

        // QKV: [M,1536] bf16
        gemm_bf16_kernel<<<dim3(12, 128), 256, 0, stream>>>(
            bufB, wq_bf, bqkv + (size_t)g * 3 * kD, nullptr, bufC, M, 3 * kD, kD, 0);
        // attention -> bufB (bf16)
        attn_kernel<<<dim3(kN / 64, kB * kH), 128, 0, stream>>>(bufC, bufB, maxnode);
        // O projection -> bufD (fp32)
        gemm_bf16_kernel<<<dim3(4, 128), 256, 0, stream>>>(
            bufB, wo_bf, bo + (size_t)g * kD, bufD, nullptr, M, kD, kD, 0);
        // LN1: x1 = LN(xp + o) -> bufD fp32 (in place), bufB bf16
        ln_kernel<<<M, 128, 0, stream>>>(bufA, bufD, g1 + (size_t)g * kD, be1 + (size_t)g * kD,
                                         bufD, bufB);
        // FFN: h = relu(x1 W1^T + b1) -> bufC bf16
        gemm_bf16_kernel<<<dim3(16, 128), 256, 0, stream>>>(
            bufB, w1_bf, b1 + (size_t)g * kDFF, nullptr, bufC, M, kDFF, kD, 1);
        // f = h W2^T + b2 -> bufA fp32
        gemm_bf16_kernel<<<dim3(4, 128), 256, 0, stream>>>(
            bufC, w2_bf, b2 + (size_t)g * kD, bufA, nullptr, M, kD, kDFF, 0);
        // LN2: y = LN(x1 + f) -> bufA fp32
        ln_kernel<<<M, 128, 0, stream>>>(bufD, bufA, g2 + (size_t)g * kD, be2 + (size_t)g * kD,
                                         bufA, nullptr);
        // scatter back
        scatter_kernel<<<M, 128, 0, stream>>>(bufA, rank, out);
    }
}
